// RWKV8_ROSA_Block_20126216749887
// MI455X (gfx1250) — matrix operand order repaired and checked by reading, not compiled, not run on hardware
//
#include <hip/hip_runtime.h>

using f16 = _Float16;
typedef __attribute__((ext_vector_type(16))) _Float16 v16h;
typedef __attribute__((ext_vector_type(8)))  _Float16 v8h;
typedef __attribute__((ext_vector_type(8)))  float    v8f;

// LDS row pitch in halves: 96 bytes -> 16B/32B aligned, 24-dword bank stride
#define LDS_PAD 48

// ---------------------------------------------------------------------------
// CDNA5 async global->LDS copy (ASYNCcnt-tracked), per-lane 16B
// ---------------------------------------------------------------------------
__device__ __forceinline__ void async_cp16(uint32_t lds_off, const f16* g) {
    asm volatile("global_load_async_to_lds_b128 %0, %1, off"
                 :: "v"(lds_off), "v"(g) : "memory");
}
__device__ __forceinline__ void wait_async_all() {
    asm volatile("s_wait_asynccnt 0x0" ::: "memory");
}

// ---------------------------------------------------------------------------
// f32 -> f16 conversion
// ---------------------------------------------------------------------------
__global__ __launch_bounds__(256) void cvt_f32_f16(const float* __restrict__ in,
                                                   f16* __restrict__ out, int n) {
    int i = blockIdx.x * blockDim.x + threadIdx.x;
    if (i < n) out[i] = (f16)in[i];
}

// ---------------------------------------------------------------------------
// LayerNorm: one block per row, wave32 shfl reduction
// ---------------------------------------------------------------------------
__global__ __launch_bounds__(256) void ln_kernel(const float* __restrict__ x,
                                                 const float* __restrict__ w,
                                                 const float* __restrict__ b,
                                                 f16* __restrict__ out, int C) {
    int row = blockIdx.x;
    const float* xr = x + (size_t)row * C;
    float s = 0.f, s2 = 0.f;
    for (int i = threadIdx.x; i < C; i += blockDim.x) {
        float v = xr[i];
        s += v; s2 += v * v;
    }
    for (int off = 16; off > 0; off >>= 1) {
        s  += __shfl_xor(s,  off, 32);
        s2 += __shfl_xor(s2, off, 32);
    }
    __shared__ float sh[16];
    int wave = threadIdx.x >> 5, lane = threadIdx.x & 31;
    if (lane == 0) { sh[wave] = s; sh[8 + wave] = s2; }
    __syncthreads();
    s = 0.f; s2 = 0.f;
    #pragma unroll
    for (int i = 0; i < 8; ++i) { s += sh[i]; s2 += sh[8 + i]; }
    float mu   = s / C;
    float var  = s2 / C - mu * mu;
    float rinv = rsqrtf(var + 1e-5f);
    for (int i = threadIdx.x; i < C; i += blockDim.x) {
        float v = (xr[i] - mu) * rinv * w[i] + b[i];
        out[(size_t)row * C + i] = (f16)v;
    }
}

// ---------------------------------------------------------------------------
// WMMA GEMM: out(M,N) = A(M,K) * W(N,K)^T
// Block tile 128x64x32, 8 waves (4x2), each wave 32x32 = 2x2 WMMA accs.
// Double-buffered async global->LDS staging (ASYNCcnt), LDS-fed fragments.
// EPI 0: f32 store | EPI 1: relu^2 -> f16 | EPI 2: +residual -> f32
// ---------------------------------------------------------------------------
template <int EPI>
__global__ __launch_bounds__(256) void wmma_gemm(const f16* __restrict__ A,
                                                 const f16* __restrict__ W,
                                                 float* __restrict__ outF,
                                                 f16* __restrict__ outH,
                                                 const float* __restrict__ res,
                                                 int M, int N, int K) {
    constexpr int BM = 128, BN = 64, BK = 32;
    __shared__ __align__(32) f16 Ash[2 * BM * LDS_PAD];
    __shared__ __align__(32) f16 Bsh[2 * BN * LDS_PAD];

    const int tid    = threadIdx.x;
    const int lane   = tid & 31;
    const int wave   = tid >> 5;
    const int wm     = (wave & 3) * 32;   // wave tile row offset in block
    const int wn     = (wave >> 2) * 32;  // wave tile col offset
    const int colsel = lane >> 4;         // 0: lanes 0-15, 1: lanes 16-31
    const int l15    = lane & 15;

    const int mblk = blockIdx.x * BM;
    const int nblk = blockIdx.y * BN;

    const uint32_t a_base = (uint32_t)(uintptr_t)&Ash[0];
    const uint32_t b_base = (uint32_t)(uintptr_t)&Bsh[0];
    constexpr uint32_t ABUF = (uint32_t)sizeof(f16) * BM * LDS_PAD;
    constexpr uint32_t BBUF = (uint32_t)sizeof(f16) * BN * LDS_PAD;

    // Cooperative tile fetch: A = 128 rows x 4x16B chunks (2 per thread),
    //                         B =  64 rows x 4x16B chunks (1 per thread).
    auto issue_tile = [&](int k0, int buf) {
        const uint32_t abuf = a_base + (uint32_t)buf * ABUF;
        const uint32_t bbuf = b_base + (uint32_t)buf * BBUF;
        #pragma unroll
        for (int p = 0; p < 2; ++p) {
            int idx = tid + p * 256;
            int row = idx >> 2, ch = idx & 3;
            const f16* g = A + (size_t)(mblk + row) * K + k0 + ch * 8;
            async_cp16(abuf + (uint32_t)(row * LDS_PAD + ch * 8) * 2, g);
        }
        {
            int row = tid >> 2, ch = tid & 3;
            const f16* g = W + (size_t)(nblk + row) * K + k0 + ch * 8;
            async_cp16(bbuf + (uint32_t)(row * LDS_PAD + ch * 8) * 2, g);
        }
    };

    v8f acc[2][2] = {};
    issue_tile(0, 0);
    const int nk = K / BK;
    for (int s = 0; s < nk; ++s) {
        const int buf = s & 1;
        wait_async_all();    // my async writes (issued last iter) are done
        __syncthreads();     // everyone's writes done + everyone done reading
        if (s + 1 < nk) issue_tile((s + 1) * BK, buf ^ 1);  // overlap next fetch

        const f16* As = &Ash[buf * BM * LDS_PAD];
        const f16* Bs = &Bsh[buf * BN * LDS_PAD];

        v16h af[2], bf[2];
        #pragma unroll
        for (int i = 0; i < 2; ++i) {
            // 16-bit A 16x32: lane group holds contiguous 16 K-halves of row
            int row = wm + i * 16 + l15;
            af[i] = *(const v16h*)(As + row * LDS_PAD + colsel * 16);
        }
        #pragma unroll
        for (int j = 0; j < 2; ++j) {
            // 16-bit B 32x16: lane group holds contiguous 16 K-halves of col
            int row = wn + j * 16 + l15;
            bf[j] = *(const v16h*)(Bs + row * LDS_PAD + colsel * 16);
        }
        #pragma unroll
        for (int i = 0; i < 2; ++i)
            #pragma unroll
            for (int j = 0; j < 2; ++j)
                acc[i][j] = __builtin_amdgcn_wmma_f32_16x16x32_f16(
                    false, af[i], false, bf[j], (short)0, acc[i][j], false, false);
    }

    // D layout: VGPR r -> row +r (+8 for lanes>=16), col = lane&15
    #pragma unroll
    for (int i = 0; i < 2; ++i) {
        #pragma unroll
        for (int j = 0; j < 2; ++j) {
            const int row_d = mblk + wm + i * 16 + colsel * 8;
            const int col_d = nblk + wn + j * 16 + l15;
            #pragma unroll
            for (int r = 0; r < 8; ++r) {
                const size_t o = (size_t)(row_d + r) * N + col_d;
                float v = acc[i][j][r];
                if (EPI == 0) {
                    outF[o] = v;
                } else if (EPI == 1) {
                    float t = v > 0.f ? v : 0.f;
                    outH[o] = (f16)(t * t);
                } else {
                    outF[o] = v + res[o];
                }
            }
        }
    }
}

// ---------------------------------------------------------------------------
// wkv recurrence + sigmoid routing + residual over fused (M x 4C) projections
//   cols [0,C)=r  [C,2C)=k  [2C,3C)=v  [3C,4C)=router
// ---------------------------------------------------------------------------
__global__ __launch_bounds__(256) void scan_kernel(const float* __restrict__ x,
                                                   const float* __restrict__ qkvr,
                                                   const float* __restrict__ tm,
                                                   float* __restrict__ x1,
                                                   int B, int T, int C) {
    int idx = blockIdx.x * blockDim.x + threadIdx.x;
    if (idx >= B * C) return;
    int b = idx / C, c = idx % C;
    float a = __expf(-__expf(tm[c]));   // e^{w_c}, w_c = -exp(time_mix_w[c])
    size_t base  = (size_t)b * T * C + c;
    size_t base4 = (size_t)b * T * (4 * C) + c;
    float s = 0.f;
    for (int t = 0; t < T; ++t) {
        size_t o  = base  + (size_t)t * C;
        size_t o4 = base4 + (size_t)t * (4 * C);
        float kv = qkvr[o4 + C] * qkvr[o4 + 2 * C];
        s = a * s + kv;
        float sig = 1.f / (1.f + __expf(-qkvr[o4 + 3 * C]));
        x1[o] = x[o] + qkvr[o4] * (s * sig);
    }
}

// ---------------------------------------------------------------------------
// Host driver
// ---------------------------------------------------------------------------
extern "C" void kernel_launch(void* const* d_in, const int* in_sizes, int n_in,
                              void* d_out, int out_size, void* d_ws, size_t ws_size,
                              hipStream_t stream) {
    const float* x    = (const float*)d_in[0];
    const float* ln1w = (const float*)d_in[1];
    const float* ln1b = (const float*)d_in[2];
    const float* Wr   = (const float*)d_in[3];
    const float* Wk   = (const float*)d_in[4];
    const float* Wv   = (const float*)d_in[5];
    const float* tm   = (const float*)d_in[6];
    const float* Wu   = (const float*)d_in[7];
    const float* ln2w = (const float*)d_in[8];
    const float* ln2b = (const float*)d_in[9];
    const float* Wck  = (const float*)d_in[10];
    const float* Wcv  = (const float*)d_in[11];
    float* out = (float*)d_out;

    constexpr int B = 8, T = 512, C = 768, C4 = 3072;
    constexpr int M = B * T;  // 4096

    char* cur = (char*)d_ws;
    auto alloc = [&](size_t bytes) -> char* {
        char* p = cur;
        cur += (bytes + 255) & ~(size_t)255;
        return p;
    };
    f16*   wqkvr_h = (f16*)alloc((size_t)C4 * C * sizeof(f16));  // [r;k;v;u] rows
    f16*   wck_h   = (f16*)alloc((size_t)C4 * C * sizeof(f16));
    f16*   wcv_h   = (f16*)alloc((size_t)C * C4 * sizeof(f16));
    f16*   xx_h    = (f16*)alloc((size_t)M * C * sizeof(f16));
    f16*   xx2_h   = (f16*)alloc((size_t)M * C * sizeof(f16));
    f16*   kcm_h   = (f16*)alloc((size_t)M * C4 * sizeof(f16));
    float* qkvr    = (float*)alloc((size_t)M * C4 * sizeof(float));
    float* x1      = (float*)alloc((size_t)M * C * sizeof(float));

    // 1) weight conversion (pack r/k/v/router contiguously -> one fused GEMM)
    const int nCC = C * C;
    cvt_f32_f16<<<(nCC + 255) / 256, 256, 0, stream>>>(Wr, wqkvr_h + 0 * nCC, nCC);
    cvt_f32_f16<<<(nCC + 255) / 256, 256, 0, stream>>>(Wk, wqkvr_h + 1 * nCC, nCC);
    cvt_f32_f16<<<(nCC + 255) / 256, 256, 0, stream>>>(Wv, wqkvr_h + 2 * nCC, nCC);
    cvt_f32_f16<<<(nCC + 255) / 256, 256, 0, stream>>>(Wu, wqkvr_h + 3 * nCC, nCC);
    cvt_f32_f16<<<(C4 * C + 255) / 256, 256, 0, stream>>>(Wck, wck_h, C4 * C);
    cvt_f32_f16<<<(C * C4 + 255) / 256, 256, 0, stream>>>(Wcv, wcv_h, C * C4);

    // 2) LN1
    ln_kernel<<<M, 256, 0, stream>>>(x, ln1w, ln1b, xx_h, C);

    // 3) fused r/k/v/router projection (WMMA, N = 4C)
    dim3 gQ(M / 128, C4 / 64);
    wmma_gemm<0><<<gQ, 256, 0, stream>>>(xx_h, wqkvr_h, qkvr, nullptr, nullptr,
                                         M, C4, C);

    // 4) wkv scan + routing + residual
    scan_kernel<<<(B * C + 255) / 256, 256, 0, stream>>>(x, qkvr, tm, x1, B, T, C);

    // 5) LN2
    ln_kernel<<<M, 256, 0, stream>>>(x1, ln2w, ln2b, xx2_h, C);

    // 6) channel-mix up-projection with relu^2 (WMMA)
    dim3 gU(M / 128, C4 / 64);
    wmma_gemm<1><<<gU, 256, 0, stream>>>(xx2_h, wck_h, nullptr, kcm_h, nullptr,
                                         M, C4, C);

    // 7) channel-mix down-projection + residual (WMMA)
    dim3 gD(M / 128, C / 64);
    wmma_gemm<2><<<gD, 256, 0, stream>>>(kcm_h, wcv_h, out, nullptr, x1,
                                         M, C, C4);
}
